// CommittorNetBP_27891517620661
// MI455X (gfx1250) — matrix operand order repaired and checked by reading, not compiled, not run on hardware
//
#include <hip/hip_runtime.h>
#include <hip/hip_fp16.h>

#define NP 512
#define NNODES 256
#define BATCH 128
#define BOXF 10.0f
#define INV_BOX 0.1f
#define RCV 2.5f
#define RC2 6.25f
#define PI_F 3.1415927410125732f

typedef __attribute__((ext_vector_type(16))) _Float16 v16h;
typedef __attribute__((ext_vector_type(8)))  float    v8f;

// ---------------- Kernel 0: W1 fp32 -> f16 (layout preserved: [n][k]) ----------------
__global__ void cvt_w1_kernel(const float* __restrict__ W1, _Float16* __restrict__ W1h) {
    int idx = blockIdx.x * blockDim.x + threadIdx.x;
    if (idx < NNODES * NP) W1h[idx] = (_Float16)W1[idx];
}

// ---------------- Kernel 1: pairwise envelope sum -> Ah [B][NP] f16 ----------------
// grid (BATCH, 2), block 256: block (b, z) handles i in [z*256, z*256+256)
__global__ void pair_env_kernel(const float* __restrict__ x, _Float16* __restrict__ Ah) {
    __shared__ float sx[NP];
    __shared__ float sy[NP];
    __shared__ float sz[NP];
    const int b = blockIdx.x;
    const int t = threadIdx.x;
    const float* xb = x + (size_t)b * NP * 3;
    for (int i = t; i < NP; i += 256) {
        sx[i] = xb[3 * i + 0];
        sy[i] = xb[3 * i + 1];
        sz[i] = xb[3 * i + 2];
    }
    __syncthreads();
    const int i = blockIdx.y * 256 + t;
    const float xi = sx[i], yi = sy[i], zi = sz[i];
    float acc = 0.0f;
    for (int j = 0; j < NP; ++j) {
        float dx = xi - sx[j]; dx -= rintf(dx * INV_BOX) * BOXF;
        float dy = yi - sy[j]; dy -= rintf(dy * INV_BOX) * BOXF;
        float dz = zi - sz[j]; dz -= rintf(dz * INV_BOX) * BOXF;
        float d2 = dx * dx + dy * dy + dz * dz;
        if (d2 > 0.0f && d2 < RC2) {
            float d   = __fsqrt_rn(d2);
            float env = __expf(-d2) * 0.5f * (__cosf((PI_F / RCV) * d) + 1.0f);
            acc += env;
        }
    }
    Ah[(size_t)b * NP + i] = (_Float16)acc;
}

// ---------------- Kernel 2: WMMA GEMM H = relu(A @ W1^T + b1) ----------------
// A: [128 x 512] f16 row-major; B fragment from W1h [n][k] (k contiguous).
// grid (N/16=16, M/16=8), block 32 (one full wave32 per 16x16 tile; EXEC all 1s).
__global__ void __launch_bounds__(32)
gemm_wmma_kernel(const _Float16* __restrict__ Ah,
                 const _Float16* __restrict__ W1h,
                 const float* __restrict__ b1,
                 float* __restrict__ H) {
    const int lane = threadIdx.x;
    const int lo = lane & 15;
    const int hi = lane >> 4;
    const int mrow = blockIdx.y * 16 + lo;   // A row held by this lane
    const int ncol = blockIdx.x * 16 + lo;   // B column held by this lane

    const unsigned int* Arow = (const unsigned int*)(Ah  + (size_t)mrow * NP);
    const unsigned int* Brow = (const unsigned int*)(W1h + (size_t)ncol * NP);

    v8f c = {};
    for (int k0 = 0; k0 < NP; k0 += 32) {
        union { v16h v; unsigned int u[8]; } a, bb;
        const int kp = k0 >> 1;  // dword (f16-pair) index of K-block base
        #pragma unroll
        for (int r = 0; r < 8; ++r) {
            // A k order used here (contiguous, not AMD table order): VGPR r <- K = 2r,2r+1 (+16 for lanes 16-31)
            const int pa = kp + r + 8 * hi;
            a.u[r] = Arow[pa];
            // B k order used here (same contiguous order as A): VGPR r <- K = 2r,2r+1 (+16 for lanes 16-31)
            bb.u[r] = Brow[kp + r + 8 * hi];
        }
        c = __builtin_amdgcn_wmma_f32_16x16x32_f16(false, a.v, false, bb.v,
                                                   (short)0, c, false, false);
    }

    // Epilogue: bias + ReLU.  D layout: n = lane%16, m = r + 8*hi.
    const float bias = b1[ncol];
    #pragma unroll
    for (int r = 0; r < 8; ++r) {
        const int m = blockIdx.y * 16 + r + 8 * hi;
        float v = c[r] + bias;
        H[(size_t)m * NNODES + ncol] = v > 0.0f ? v : 0.0f;
    }
}

// ---------------- Kernel 3: out = sigmoid(H @ W2^T) ----------------
__global__ void head_kernel(const float* __restrict__ H, const float* __restrict__ W2,
                            float* __restrict__ out) {
    __shared__ float w2s[NNODES];
    const int t = threadIdx.x;
    for (int n = t; n < NNODES; n += 128) w2s[n] = W2[n];
    __syncthreads();
    if (t < BATCH) {
        const float* hr = H + (size_t)t * NNODES;
        float s = 0.0f;
        #pragma unroll 8
        for (int n = 0; n < NNODES; ++n) s += hr[n] * w2s[n];
        out[t] = 1.0f / (1.0f + __expf(-s));
    }
}

extern "C" void kernel_launch(void* const* d_in, const int* in_sizes, int n_in,
                              void* d_out, int out_size, void* d_ws, size_t ws_size,
                              hipStream_t stream) {
    const float* x  = (const float*)d_in[0];   // [128, 512, 3]
    const float* W1 = (const float*)d_in[1];   // [256, 512]
    const float* b1 = (const float*)d_in[2];   // [256]
    const float* W2 = (const float*)d_in[3];   // [1, 256]
    float* out = (float*)d_out;                // [128, 1]

    char* ws = (char*)d_ws;
    _Float16* W1h = (_Float16*)(ws);                      // 256*512*2 = 256 KB
    _Float16* Ah  = (_Float16*)(ws + 262144);             // 128*512*2 = 128 KB
    float*    H   = (float*)(ws + 262144 + 131072);       // 128*256*4 = 128 KB
    (void)in_sizes; (void)n_in; (void)out_size; (void)ws_size;

    hipLaunchKernelGGL(cvt_w1_kernel,   dim3(512),        dim3(256), 0, stream, W1, W1h);
    hipLaunchKernelGGL(pair_env_kernel, dim3(BATCH, 2),   dim3(256), 0, stream, x, Ah);
    hipLaunchKernelGGL(gemm_wmma_kernel,dim3(16, 8),      dim3(32),  0, stream, Ah, W1h, b1, H);
    hipLaunchKernelGGL(head_kernel,     dim3(1),          dim3(128), 0, stream, H, W2, out);
}
